// MultiHeadAttention_76098230550845
// MI455X (gfx1250) — matrix operand order repaired and checked by reading, not compiled, not run on hardware
//
#include <hip/hip_runtime.h>

#define EMB   128
#define HEADS 8
#define BATCH 4
#define TSEQ  2048
#define MTOT  (BATCH * TSEQ)          // 8192 rows (b*T+t)
#define FDIM  (HEADS * EMB)           // 1024
#define WELEM (FDIM * EMB)            // 131072 per weight matrix
#define HEADELEM ((size_t)BATCH * HEADS * TSEQ * EMB)  // 8388608

typedef __attribute__((ext_vector_type(16))) _Float16 v16h;
typedef __attribute__((ext_vector_type(8)))  float    v8f;
typedef unsigned int uint32x4 __attribute__((ext_vector_type(4)));
typedef int          int32x8  __attribute__((ext_vector_type(8)));
typedef int          int32x4  __attribute__((ext_vector_type(4)));

__device__ __forceinline__ v8f vzero8() {
    v8f z;
#pragma unroll
    for (int i = 0; i < 8; ++i) z[i] = 0.0f;
    return z;
}

__device__ __forceinline__ v8f wmma_f16(v16h a, v16h b, v8f c) {
    return __builtin_amdgcn_wmma_f32_16x16x32_f16(false, a, false, b, (short)0, c, false, false);
}

// ---- TDM: 2D tile load Global -> LDS (f16 elements). Dims/strides in elements,
// global addr in bytes, per CDNA5 ISA ch.8 D# layout. Issue from ONE wave only.
__device__ __forceinline__ void tdm_load_2d_f16(unsigned lds_off, unsigned long long gaddr,
                                                unsigned dim0, unsigned dim1,
                                                unsigned tile0, unsigned tile1,
                                                unsigned long long stride0) {
    unsigned long long s1 = stride0 * (unsigned long long)dim1;
    uint32x4 g0;
    g0[0] = 1u;                                              // count=1, user descriptor
    g0[1] = lds_off;                                         // lds_addr (bytes)
    g0[2] = (unsigned)(gaddr & 0xFFFFFFFFull);               // global_addr[31:0]
    g0[3] = (unsigned)((gaddr >> 32) & 0x01FFFFFFull)        // global_addr[56:32]
          | 0x80000000u;                                     // type=2 (bits 127:126 = 10b)
    int32x8 g1;
    g1[0] = (int)(1u << 16);                                 // data_size=1 -> 2 bytes
    g1[1] = (int)((dim0 & 0xFFFFu) << 16);                   // tensor_dim0[15:0]
    g1[2] = (int)((dim0 >> 16) | ((dim1 & 0xFFFFu) << 16));  // dim0[31:16] | dim1[15:0]
    g1[3] = (int)((dim1 >> 16) | (tile0 << 16));             // dim1[31:16] | tile_dim0
    g1[4] = (int)(tile1 & 0xFFFFu);                          // tile_dim1 (tile_dim2=0)
    g1[5] = (int)(stride0 & 0xFFFFFFFFull);                  // dim0_stride[31:0]
    g1[6] = (int)((unsigned)(stride0 >> 32) | ((unsigned)(s1 & 0xFFFFull) << 16));
    g1[7] = (int)((unsigned)(s1 >> 16));                     // dim1_stride[47:16]
    int32x4 z4 = {0, 0, 0, 0};
    int32x8 z8 = {0, 0, 0, 0, 0, 0, 0, 0};
    __builtin_amdgcn_tensor_load_to_lds(g0, g1, z4, z4, z8, 0);
}

// ---- Fragment loaders per CDNA5 ISA 7.12.2 (wave32) ----
__device__ __forceinline__ v16h load_a_frag(const _Float16* __restrict__ src, int ld) {
    int lane = threadIdx.x & 31;
    int m = lane & 15;
    int koff = (lane >> 4) << 3;  // 0 or 8
    v16h a;
#pragma unroll
    for (int j = 0; j < 8; ++j) {
        int k = (j < 4) ? (koff + 2 * j) : (16 + koff + 2 * (j - 4));
        a[2 * j]     = src[m * ld + k];
        a[2 * j + 1] = src[m * ld + k + 1];
    }
    return a;
}

__device__ __forceinline__ v16h load_a_frag_f32(const float* __restrict__ src, int ld) {
    int lane = threadIdx.x & 31;
    int m = lane & 15;
    int koff = (lane >> 4) << 3;
    v16h a;
#pragma unroll
    for (int j = 0; j < 8; ++j) {
        int k = (j < 4) ? (koff + 2 * j) : (16 + koff + 2 * (j - 4));
        a[2 * j]     = (_Float16)src[m * ld + k];
        a[2 * j + 1] = (_Float16)src[m * ld + k + 1];
    }
    return a;
}

// B 32x16 from NxK row-major source (B[k][n] = src[n*ld + k]) -> table k order (matches A)
__device__ __forceinline__ v16h load_b_frag_t(const _Float16* __restrict__ src, int ld) {
    int lane = threadIdx.x & 31;
    int n = lane & 15;
    int koff = (lane >> 4) << 3;
    v16h b;
#pragma unroll
    for (int j = 0; j < 8; ++j) {
        int k = (j < 4) ? (koff + 2 * j) : (16 + koff + 2 * (j - 4));
        b[2 * j]     = src[n * ld + k];
        b[2 * j + 1] = src[n * ld + k + 1];
    }
    return b;
}

// ---- Kernel 0: cast the four weight matrices fp32 -> f16 into workspace ----
__global__ void cast_weights_kernel(const float* __restrict__ Wq, const float* __restrict__ Wk,
                                    const float* __restrict__ Wv, const float* __restrict__ Wu,
                                    _Float16* __restrict__ out) {
    int i = blockIdx.x * blockDim.x + threadIdx.x;
    if (i < WELEM)              out[i] = (_Float16)Wq[i];
    else if (i < 2 * WELEM)     out[i] = (_Float16)Wk[i - WELEM];
    else if (i < 3 * WELEM)     out[i] = (_Float16)Wv[i - 2 * WELEM];
    else if (i < 4 * WELEM)     out[i] = (_Float16)Wu[i - 3 * WELEM];
}

// ---- Kernel 1: QKV projection GEMM. grid = (M/16, F/64, 3), block = 32 (1 wave)
// which 0/1 (q,k): Y[b,h,t,e]; which 2 (v): transposed Y[b,h,e,t] for TDM/LDS use
__global__ void proj_kernel(const float* __restrict__ q, const float* __restrict__ k,
                            const float* __restrict__ v, const _Float16* __restrict__ Wh,
                            _Float16* __restrict__ Ybase) {
    const int which = blockIdx.z;
    const float* X = (which == 0) ? q : ((which == 1) ? k : v);
    const _Float16* W = Wh + (size_t)which * WELEM;
    _Float16* Y = Ybase + (size_t)which * HEADELEM;

    const int row0 = blockIdx.x * 16;  // token index b*T+t
    const int col0 = blockIdx.y * 64;  // f index

    v16h a[4];
#pragma unroll
    for (int c = 0; c < 4; ++c)
        a[c] = load_a_frag_f32(X + (size_t)row0 * EMB + c * 32, EMB);

    v8f acc[4];
#pragma unroll
    for (int nt = 0; nt < 4; ++nt) acc[nt] = vzero8();

#pragma unroll
    for (int nt = 0; nt < 4; ++nt) {
#pragma unroll
        for (int c = 0; c < 4; ++c) {
            v16h b = load_b_frag_t(W + (size_t)(col0 + nt * 16) * EMB + c * 32, EMB);
            acc[nt] = wmma_f16(a[c], b, acc[nt]);
        }
    }

    const int lane = threadIdx.x & 31;
    const int n = lane & 15;
    const int mbase = (lane >> 4) << 3;
#pragma unroll
    for (int nt = 0; nt < 4; ++nt) {
#pragma unroll
        for (int r = 0; r < 8; ++r) {
            int m = row0 + mbase + r;          // b*T + t
            int f = col0 + nt * 16 + n;        // h*E + e
            int b_ = m >> 11, t = m & (TSEQ - 1);
            int h = f >> 7, e = f & (EMB - 1);
            size_t head = (size_t)(b_ * HEADS + h);
            if (which == 2)
                Y[(head * EMB + e) * TSEQ + t] = (_Float16)acc[nt][r];   // (b,h,e,t)
            else
                Y[(head * TSEQ + t) * EMB + e] = (_Float16)acc[nt][r];   // (b,h,t,e)
        }
    }
}

// ---- Kernel 2: flash attention. grid = (B*H, T/64), block = 128 (4 waves)
// kh tiles arrive row-major (key,e); vh tiles arrive TRANSPOSED (e,key) via TDM.
__global__ void attn_kernel(const _Float16* __restrict__ qh, const _Float16* __restrict__ kh,
                            const _Float16* __restrict__ vh, _Float16* __restrict__ O) {
    __shared__ __align__(16) _Float16 Kt[32 * EMB];     // (key, e)
    __shared__ __align__(16) _Float16 Vt[EMB * 32];     // (e, key)  -- transposed
    __shared__ __align__(16) _Float16 Ps[4][16 * 32];

    const int bh = blockIdx.x;       // b*H + h
    const int qt = blockIdx.y;       // query tile of 64
    const int wave = threadIdx.x >> 5;
    const int lane = threadIdx.x & 31;
    const int ncol = lane & 15;
    const int mbase = (lane >> 4) << 3;

    const _Float16* Qb = qh + (size_t)bh * TSEQ * EMB;
    const _Float16* Kb = kh + (size_t)bh * TSEQ * EMB;
    const _Float16* Vb = vh + (size_t)bh * EMB * TSEQ;   // transposed layout (e,t)

    const unsigned ldsK = (unsigned)(uintptr_t)(&Kt[0]);
    const unsigned ldsV = (unsigned)(uintptr_t)(&Vt[0]);

    const int qrow0 = qt * 64 + wave * 16;

    v16h aq[4];
#pragma unroll
    for (int c = 0; c < 4; ++c)
        aq[c] = load_a_frag(Qb + (size_t)qrow0 * EMB + c * 32, EMB);

    v8f o[8];
#pragma unroll
    for (int j = 0; j < 8; ++j) o[j] = vzero8();
    float mrow[8], lrow[8];
#pragma unroll
    for (int r = 0; r < 8; ++r) { mrow[r] = -1e30f; lrow[r] = 0.0f; }

    const float scale = 0.08838834764831845f;  // 1/sqrt(EMB)
    const int nkt = (qt * 64 + 63) / 32 + 1;   // causal bound for this WG (key tiles of 32)

    for (int kt = 0; kt < nkt; ++kt) {
        const int krow0 = kt * 32;
        __syncthreads();   // previous tile fully consumed
        if (wave == 0) {
            // K tile: 32 rows of (T,E) tensor -> LDS row-major (key,e)
            tdm_load_2d_f16(ldsK,
                            (unsigned long long)(uintptr_t)(Kb + (size_t)krow0 * EMB),
                            EMB, TSEQ, EMB, 32, EMB);
            // V tile: 32 cols of (E,T) tensor -> LDS (e,key) = V^T
            tdm_load_2d_f16(ldsV,
                            (unsigned long long)(uintptr_t)(Vb + (size_t)krow0),
                            TSEQ, EMB, 32, EMB, TSEQ);
            __builtin_amdgcn_s_wait_tensorcnt((short)0);
        }
        if (kt + 1 < nkt)
            __builtin_prefetch(Kb + (size_t)(krow0 + 32) * EMB, 0, 0);
        __syncthreads();   // tiles visible to all waves

        // S = Q * K^T  (16 x 32 scores as two 16x16 D-frags)
        v8f s0 = vzero8(), s1 = vzero8();
#pragma unroll
        for (int c = 0; c < 4; ++c) {
            v16h bk0 = load_b_frag_t(Kt + 0 * EMB + c * 32, EMB);
            s0 = wmma_f16(aq[c], bk0, s0);
            v16h bk1 = load_b_frag_t(Kt + 16 * EMB + c * 32, EMB);
            s1 = wmma_f16(aq[c], bk1, s1);
        }

        // online softmax, row-wise (each row lives in one 16-lane half)
#pragma unroll
        for (int r = 0; r < 8; ++r) {
            const int qrow = qrow0 + mbase + r;
            float v0 = s0[r] * scale;
            float v1 = s1[r] * scale;
            if (krow0 + ncol > qrow)      v0 = -1e30f;
            if (krow0 + 16 + ncol > qrow) v1 = -1e30f;
            float tmax = fmaxf(v0, v1);
#pragma unroll
            for (int off = 1; off < 16; off <<= 1)
                tmax = fmaxf(tmax, __shfl_xor(tmax, off, 32));
            const float mnew = fmaxf(mrow[r], tmax);
            const float corr = __expf(mrow[r] - mnew);
            const float p0 = __expf(v0 - mnew);
            const float p1 = __expf(v1 - mnew);
            float rsum = p0 + p1;
#pragma unroll
            for (int off = 1; off < 16; off <<= 1)
                rsum += __shfl_xor(rsum, off, 32);
            lrow[r] = lrow[r] * corr + rsum;
            mrow[r] = mnew;
#pragma unroll
            for (int j = 0; j < 8; ++j) o[j][r] *= corr;
            Ps[wave][(mbase + r) * 32 + ncol]      = (_Float16)p0;
            Ps[wave][(mbase + r) * 32 + 16 + ncol] = (_Float16)p1;
        }

        // O += P * V : B-frag from V^T tile, contiguous along k per lane
        v16h ap = load_a_frag(&Ps[wave][0], 32);
#pragma unroll
        for (int j = 0; j < 8; ++j) {
            v16h bv = load_b_frag_t(Vt + (j * 16) * 32, 32);
            o[j] = wmma_f16(ap, bv, o[j]);
        }
    }

    // epilogue: normalize and store to (b, t, h*E+e) f16
    const int b_ = bh >> 3;
    const int h = bh & (HEADS - 1);
#pragma unroll
    for (int j = 0; j < 8; ++j) {
#pragma unroll
        for (int r = 0; r < 8; ++r) {
            int t = qrow0 + mbase + r;
            int e = j * 16 + ncol;
            float val = o[j][r] / lrow[r];
            O[((size_t)(b_ * TSEQ + t)) * FDIM + h * EMB + e] = (_Float16)val;
        }
    }
}

// ---- Kernel 3: output projection + bias. grid = (M/16, 128/64), block = 32
__global__ void outproj_kernel(const _Float16* __restrict__ Oh, const _Float16* __restrict__ Wuh,
                               const float* __restrict__ bu, float* __restrict__ out) {
    const int row0 = blockIdx.x * 16;
    const int col0 = blockIdx.y * 64;

    v8f acc[4];
#pragma unroll
    for (int nt = 0; nt < 4; ++nt) acc[nt] = vzero8();

    for (int kc = 0; kc < FDIM / 32; ++kc) {
        v16h a = load_a_frag(Oh + (size_t)row0 * FDIM + kc * 32, FDIM);
#pragma unroll
        for (int nt = 0; nt < 4; ++nt) {
            v16h b = load_b_frag_t(Wuh + (size_t)(col0 + nt * 16) * FDIM + kc * 32, FDIM);
            acc[nt] = wmma_f16(a, b, acc[nt]);
        }
    }

    const int lane = threadIdx.x & 31;
    const int n = lane & 15;
    const int mbase = (lane >> 4) << 3;
#pragma unroll
    for (int nt = 0; nt < 4; ++nt) {
#pragma unroll
        for (int r = 0; r < 8; ++r) {
            int e = col0 + nt * 16 + n;
            out[(size_t)(row0 + mbase + r) * EMB + e] = acc[nt][r] + bu[e];
        }
    }
}

extern "C" void kernel_launch(void* const* d_in, const int* in_sizes, int n_in,
                              void* d_out, int out_size, void* d_ws, size_t ws_size,
                              hipStream_t stream) {
    const float* q  = (const float*)d_in[0];
    const float* k  = (const float*)d_in[1];
    const float* v  = (const float*)d_in[2];
    const float* Wq = (const float*)d_in[3];
    const float* Wk = (const float*)d_in[4];
    const float* Wv = (const float*)d_in[5];
    const float* Wu = (const float*)d_in[6];
    const float* bu = (const float*)d_in[7];
    float* out = (float*)d_out;

    char* ws = (char*)d_ws;
    _Float16* Wh = (_Float16*)ws;                                  // 4*WELEM f16 = 1 MB
    _Float16* qkv = (_Float16*)(ws + (size_t)(1 << 20));           // 3 * 16 MB
    _Float16* Oh  = qkv + 3 * HEADELEM;                            // 16 MB

    cast_weights_kernel<<<(4 * WELEM + 255) / 256, 256, 0, stream>>>(Wq, Wk, Wv, Wu, Wh);

    dim3 g1(MTOT / 16, FDIM / 64, 3);
    proj_kernel<<<g1, 32, 0, stream>>>(q, k, v, Wh, qkv);

    dim3 g2(BATCH * HEADS, TSEQ / 64);
    attn_kernel<<<g2, 128, 0, stream>>>(qkv, qkv + HEADELEM, qkv + 2 * HEADELEM, Oh);

    dim3 g3(MTOT / 16, EMB / 64);
    outproj_kernel<<<g3, 32, 0, stream>>>(Oh, Wh + 3 * (size_t)WELEM, bu, out);
}
